// AttentionLayer_11940009083374
// MI455X (gfx1250) — matrix operand order repaired and checked by reading, not compiled, not run on hardware
//
#include <hip/hip_runtime.h>
#include <math.h>

#define S_LEN 4096
#define BATCH 4
#define DH    128
#define TS    64                 // keys per staged tile
#define NT    (S_LEN / TS)       // 64 tiles

typedef __bf16        v16bf __attribute__((ext_vector_type(16)));
typedef float         v8f   __attribute__((ext_vector_type(8)));
typedef float         f32x4 __attribute__((ext_vector_type(4)));
typedef unsigned int  u32x4 __attribute__((ext_vector_type(4)));
typedef int           i32x4 __attribute__((ext_vector_type(4)));
typedef int           i32x8 __attribute__((ext_vector_type(8)));

#if __has_builtin(__builtin_amdgcn_tensor_load_to_lds)
#define HAVE_TDM 1
#else
#define HAVE_TDM 0
#endif

// compile-probe: report (device pass only) whether the TDM builtin exists
#if defined(__AMDGCN__)
#if HAVE_TDM
#pragma message("[cdna5-probe] device pass: __builtin_amdgcn_tensor_load_to_lds AVAILABLE")
#else
#pragma message("[cdna5-probe] device pass: __builtin_amdgcn_tensor_load_to_lds MISSING (cooperative-copy fallback)")
#endif
#endif

union Frag { v16bf v; u32x4 q[2]; unsigned short s[16]; };

__device__ inline unsigned short f2bf(float f) {
    union { __bf16 b; unsigned short s; } u;
    u.b = (__bf16)f;             // native f32->bf16 convert (v_cvt_pk_bf16_f32)
    return u.s;
}

__device__ inline v8f wmma_bf16(v16bf a, v16bf b, v8f c) {
    return __builtin_amdgcn_wmma_f32_16x16x32_bf16(false, a, false, b, (short)0, c, false, false);
}

#if HAVE_TDM
__device__ inline unsigned lds_off_of(const void* p) {
    return (unsigned)(unsigned long long)(__attribute__((address_space(3))) const char*)p;
}

// 2-D TDM load: tile (tile1 rows x tile0 elems) of a (td1 x td0) bf16 tensor,
// row stride stride0 elems, packed contiguously into LDS at lds_off.
__device__ inline void tdm_load_2d(unsigned lds_off, const void* gptr,
                                   unsigned td0, unsigned td1,
                                   unsigned tile0, unsigned tile1,
                                   unsigned long long stride0)
{
    unsigned long long ga = (unsigned long long)gptr;
    u32x4 g0;
    g0[0] = 1u;                                                  // count=1, user desc
    g0[1] = lds_off;                                             // lds_addr (bytes)
    g0[2] = (unsigned)ga;                                        // global_addr[31:0]
    g0[3] = (unsigned)((ga >> 32) & 0x1FFFFFFull) | (2u << 30);  // addr[56:32] | type=2
    i32x8 g1;
    g1[0] = (int)(1u << 16);                                     // data_size=1 (2B)
    g1[1] = (int)((td0 & 0xFFFFu) << 16);                        // tensor_dim0[15:0]
    g1[2] = (int)((td0 >> 16) | ((td1 & 0xFFFFu) << 16));        // td0 hi | td1 lo
    g1[3] = (int)((td1 >> 16) | (tile0 << 16));                  // td1 hi | tile_dim0
    g1[4] = (int)(tile1 & 0xFFFFu);                              // tile_dim1 (tile_dim2=0)
    g1[5] = (int)(stride0 & 0xFFFFFFFFull);                      // dim0 stride lo
    g1[6] = (int)((stride0 >> 32) & 0xFFFFull);                  // dim0 stride hi
    g1[7] = 0;
    i32x4 z4 = {0, 0, 0, 0};
#if defined(__clang_major__) && (__clang_major__ >= 23)
    i32x8 z8 = {0, 0, 0, 0, 0, 0, 0, 0};
    __builtin_amdgcn_tensor_load_to_lds(g0, g1, z4, z4, z8, 0);
#else
    __builtin_amdgcn_tensor_load_to_lds(g0, g1, z4, z4, 0);
#endif
}
#endif

// ---------------------------------------------------------------------------
// Kernel 1: QKV projection.  y = x @ W^T + b  (bf16 out, fp32 accumulation)
// grid = (S/128, B, 3)  z: 0=Q (scaled by log2e/sqrt(D)), 1=K, 2=V^T (D x S)
// ---------------------------------------------------------------------------
__global__ __launch_bounds__(256) void qkv_kernel(
    const float* __restrict__ x,
    const float* __restrict__ Wq, const float* __restrict__ bq,
    const float* __restrict__ Wk, const float* __restrict__ bk,
    const float* __restrict__ Wv, const float* __restrict__ bv,
    unsigned short* __restrict__ qws,
    unsigned short* __restrict__ kws,
    unsigned short* __restrict__ vtws)
{
    __shared__ __align__(16) unsigned short Wl[DH * DH];   // 32 KB bf16 weights

    const int z    = blockIdx.z;
    const float* W    = (z == 0) ? Wq : (z == 1) ? Wk : Wv;
    const float* bias = (z == 0) ? bq : (z == 1) ? bk : bv;
    const int tid = threadIdx.x;

    for (int i = tid; i < DH * DH / 4; i += 256) {
        f32x4 f = ((const f32x4*)W)[i];
        unsigned short* d = &Wl[i * 4];
        d[0] = f2bf(f.x); d[1] = f2bf(f.y); d[2] = f2bf(f.z); d[3] = f2bf(f.w);
    }
    __syncthreads();

    const int w = tid >> 5, lane = tid & 31;
    const int lh = lane >> 4, ln = lane & 15;
    const int b = blockIdx.y;
    const int row0 = blockIdx.x * 128 + w * 16;

    const float* xrow = x + ((size_t)b * S_LEN + row0 + ln) * DH;
    Frag a[4];
#pragma unroll
    for (int kt = 0; kt < 4; ++kt) {
        const float* p = xrow + kt * 32 + lh * 8;
        f32x4 f0 = *(const f32x4*)(p);
        f32x4 f1 = *(const f32x4*)(p + 4);
        f32x4 f2 = *(const f32x4*)(p + 16);
        f32x4 f3 = *(const f32x4*)(p + 20);
#pragma unroll
        for (int j = 0; j < 4; ++j) {
            a[kt].s[j]      = f2bf(f0[j]);
            a[kt].s[4 + j]  = f2bf(f1[j]);
            a[kt].s[8 + j]  = f2bf(f2[j]);
            a[kt].s[12 + j] = f2bf(f3[j]);
        }
    }

    // 32 WMMAs with 2-deep software-pipelined B fragments
    v8f acc[8] = {};
    Frag bfr[2];
    auto loadW = [&](int idx, int kt, int n) {
        const unsigned short* wp = &Wl[(n * 16 + ln) * DH + kt * 32 + lh * 8];
        bfr[idx].q[0] = *(const u32x4*)(wp);
        bfr[idx].q[1] = *(const u32x4*)(wp + 16);
    };
    loadW(0, 0, 0);
#pragma unroll
    for (int i = 0; i < 32; ++i) {
        const int kt = i >> 3, n = i & 7;
        if (i + 1 < 32) loadW((i + 1) & 1, (i + 1) >> 3, (i + 1) & 7);
        acc[n] = wmma_bf16(a[kt].v, bfr[i & 1].v, acc[n]);
    }

    // Q carries 1/sqrt(D) * log2(e) so phase-2 softmax runs in exp2 domain.
    const float scale = (z == 0) ? (0.08838834764831845f * 1.4426950408889634f) : 1.0f;
    unsigned short* dst = (z == 0) ? qws : (z == 1) ? kws : vtws;
#pragma unroll
    for (int n = 0; n < 8; ++n) {
        const int col = n * 16 + ln;
        const float bv_ = bias[col];
#pragma unroll
        for (int r = 0; r < 8; ++r) {
            const int row = row0 + r + lh * 8;
            unsigned short val = f2bf((acc[n][r] + bv_) * scale);
            if (z < 2) dst[((size_t)b * S_LEN + row) * DH + col] = val;   // Q,K: (S,D)
            else       dst[((size_t)b * DH + col) * S_LEN + row] = val;   // V^T: (D,S)
        }
    }
}

// ---------------------------------------------------------------------------
// Kernel 2: flash attention.  8 waves x 16 query rows; K/V^T tiles (64 keys)
// double-buffered in LDS via TDM (tensor_load_to_lds) or cooperative copy.
// ---------------------------------------------------------------------------
__global__ __launch_bounds__(256) void attn_kernel(
    const unsigned short* __restrict__ qws,
    const unsigned short* __restrict__ kws,
    const unsigned short* __restrict__ vtws,
    float* __restrict__ out)
{
    __shared__ __align__(16) unsigned short Kt[2][TS][DH];   // 2 x 16 KB
    __shared__ __align__(16) unsigned short Vl[2][DH][TS];   // 2 x 16 KB
    __shared__ __align__(16) unsigned short P[8][16][TS];    // 16 KB exp(S) tiles

    const int tid = threadIdx.x;
    const int w = tid >> 5, lane = tid & 31;
    const int lh = lane >> 4, ln = lane & 15;
    const int b = blockIdx.y;
    const int row0 = blockIdx.x * 128 + w * 16;

    const unsigned short* kbase = kws  + (size_t)b * S_LEN * DH;
    const unsigned short* vbase = vtws + (size_t)b * DH * S_LEN;

    // Q A-fragments resident for the whole key loop
    const unsigned short* qrow = qws + ((size_t)b * S_LEN + row0 + ln) * DH;
    Frag qa[4];
#pragma unroll
    for (int kt = 0; kt < 4; ++kt) {
        const unsigned short* p = qrow + kt * 32 + lh * 8;
        qa[kt].q[0] = *(const u32x4*)(p);
        qa[kt].q[1] = *(const u32x4*)(p + 16);
    }

    v8f o[8] = {};
    float mrun[8], lrun[8];
#pragma unroll
    for (int r = 0; r < 8; ++r) { mrun[r] = -3.0e38f; lrun[r] = 0.0f; }

    auto stage = [&](int buf, int t0) {
#if HAVE_TDM
        if (w == 0) {
            // K tile: 64 rows x 128 elems of (S x D) tensor
            tdm_load_2d(lds_off_of(&Kt[buf][0][0]), kbase + (size_t)t0 * DH,
                        DH, S_LEN, DH, TS, DH);
            // V tile: 128 rows x 64 elems of (D x S) tensor
            tdm_load_2d(lds_off_of(&Vl[buf][0][0]), vbase + t0,
                        S_LEN, DH, TS, DH, S_LEN);
        }
#else
        const u32x4* ksrc = (const u32x4*)(kbase + (size_t)t0 * DH);
        u32x4* kdst = (u32x4*)&Kt[buf][0][0];
#pragma unroll
        for (int i = 0; i < 4; ++i) kdst[tid + i * 256] = ksrc[tid + i * 256];
#pragma unroll
        for (int i = 0; i < 4; ++i) {
            int j = tid + i * 256;
            int row = j >> 3, c = j & 7;
            ((u32x4*)&Vl[buf][row][0])[c] =
                ((const u32x4*)(vbase + (size_t)row * S_LEN + t0))[c];
        }
#endif
    };

    stage(0, 0);

    for (int t = 0; t < NT; ++t) {
        const int buf = t & 1;
        if (t + 1 < NT) stage(buf ^ 1, (t + 1) * TS);   // prefetch next tile
#if HAVE_TDM
        if (w == 0) {
            if (t + 1 < NT) __builtin_amdgcn_s_wait_tensorcnt(2);  // tile t done
            else            __builtin_amdgcn_s_wait_tensorcnt(0);
        }
#endif
        __syncthreads();   // tile t visible to all waves

        // ---- scores: 16 query rows x 64 keys, pipelined K fragments --------
        v8f s[4] = {};
        {
            Frag kb[2];
            auto loadK = [&](int idx, int j, int dt) {
                const unsigned short* kp = &Kt[buf][j * 16 + ln][dt * 32 + lh * 8];
                kb[idx].q[0] = *(const u32x4*)(kp);
                kb[idx].q[1] = *(const u32x4*)(kp + 16);
            };
            loadK(0, 0, 0);
#pragma unroll
            for (int i = 0; i < 16; ++i) {
                const int j = i >> 2, dt = i & 3;
                if (i + 1 < 16) loadK((i + 1) & 1, (i + 1) >> 2, (i + 1) & 3);
                s[j] = wmma_bf16(qa[dt].v, kb[i & 1].v, s[j]);
            }
        }

        // ---- online softmax (exp2 domain; row r spans 16 lanes) ------------
#pragma unroll
        for (int r = 0; r < 8; ++r) {
            float vmax = fmaxf(fmaxf(s[0][r], s[1][r]), fmaxf(s[2][r], s[3][r]));
#pragma unroll
            for (int off = 8; off > 0; off >>= 1)
                vmax = fmaxf(vmax, __shfl_xor(vmax, off, 16));
            const float mn = fmaxf(mrun[r], vmax);
            const float sc = exp2f(mrun[r] - mn);
            const float p0 = exp2f(s[0][r] - mn);
            const float p1 = exp2f(s[1][r] - mn);
            const float p2 = exp2f(s[2][r] - mn);
            const float p3 = exp2f(s[3][r] - mn);
            float rs = (p0 + p1) + (p2 + p3);
#pragma unroll
            for (int off = 8; off > 0; off >>= 1)
                rs += __shfl_xor(rs, off, 16);
            lrun[r] = lrun[r] * sc + rs;
            mrun[r] = mn;
#pragma unroll
            for (int n = 0; n < 8; ++n) o[n][r] *= sc;
            const int prow = r + lh * 8;
            P[w][prow][ln]      = f2bf(p0);
            P[w][prow][ln + 16] = f2bf(p1);
            P[w][prow][ln + 32] = f2bf(p2);
            P[w][prow][ln + 48] = f2bf(p3);
        }

        // ---- P @ V : pipelined V fragments ---------------------------------
        Frag pa[2];
        {
            const unsigned short* pp = &P[w][ln][lh * 8];
            pa[0].q[0] = *(const u32x4*)(pp);
            pa[0].q[1] = *(const u32x4*)(pp + 16);
            pa[1].q[0] = *(const u32x4*)(pp + 32);
            pa[1].q[1] = *(const u32x4*)(pp + 48);
        }
        {
            Frag vb[2];
            auto loadV = [&](int idx, int n, int h) {
                const unsigned short* vp = &Vl[buf][n * 16 + ln][lh * 8 + h * 32];
                vb[idx].q[0] = *(const u32x4*)(vp);
                vb[idx].q[1] = *(const u32x4*)(vp + 16);
            };
            loadV(0, 0, 0);
#pragma unroll
            for (int i = 0; i < 16; ++i) {
                const int n = i >> 1, h = i & 1;
                if (i + 1 < 16) loadV((i + 1) & 1, (i + 1) >> 1, (i + 1) & 1);
                o[n] = wmma_bf16(pa[h].v, vb[i & 1].v, o[n]);
            }
        }
        __syncthreads();   // all waves done with tile t before buffer reuse
    }

    // ---- normalize (v_rcp) and store fp32 (B, S*D) -------------------------
#pragma unroll
    for (int r = 0; r < 8; ++r) {
        const float inv = __builtin_amdgcn_rcpf(lrun[r]);
        const int row = row0 + r + lh * 8;
        float* orow = out + ((size_t)b * S_LEN + row) * DH;
#pragma unroll
        for (int n = 0; n < 8; ++n)
            orow[n * 16 + ln] = o[n][r] * inv;
    }
}

// ---------------------------------------------------------------------------
extern "C" void kernel_launch(void* const* d_in, const int* in_sizes, int n_in,
                              void* d_out, int out_size, void* d_ws, size_t ws_size,
                              hipStream_t stream) {
    (void)in_sizes; (void)n_in; (void)out_size; (void)ws_size;
    const float* x  = (const float*)d_in[0];
    const float* Wq = (const float*)d_in[1];
    const float* bq = (const float*)d_in[2];
    const float* Wk = (const float*)d_in[3];
    const float* bk = (const float*)d_in[4];
    const float* Wv = (const float*)d_in[5];
    const float* bv = (const float*)d_in[6];

    const size_t E = (size_t)BATCH * S_LEN * DH;     // 2,097,152 elements
    unsigned short* qws = (unsigned short*)d_ws;     // bf16 Q   (B,S,D)
    unsigned short* kws = qws + E;                   // bf16 K   (B,S,D)
    unsigned short* vt  = kws + E;                   // bf16 V^T (B,D,S)  total 12 MB

    dim3 g1(S_LEN / 128, BATCH, 3);
    qkv_kernel<<<g1, 256, 0, stream>>>(x, Wq, bq, Wk, bk, Wv, bv, qws, kws, vt);

    dim3 g2(S_LEN / 128, BATCH);
    attn_kernel<<<g2, 256, 0, stream>>>(qws, kws, vt, (float*)d_out);
}
